// Grok1SparseMoeBlock_23742579212766
// MI455X (gfx1250) — matrix operand order repaired and checked by reading, not compiled, not run on hardware
//
#include <hip/hip_runtime.h>
#include <math.h>

#define DI __device__ __forceinline__

typedef __bf16 bf16_t;
typedef __attribute__((ext_vector_type(16))) __bf16 v16bf;
typedef __attribute__((ext_vector_type(8)))  float  v8f;

// ---- problem sizes ----
constexpr int Bc = 4, Sc = 2048, Hc = 1024, Ic = 4096, Ec = 8;
constexpr int Tc = Bc * Sc;          // 8192 tokens
// ---- tiling ----
constexpr int MT = 64;               // tokens per workgroup tile
constexpr int IC = 128;              // I-chunk per fused step (8 N-tiles)
constexpr int KS = 128;              // K rows of w_in/w_v staged per round
constexpr int NTHREADS = 512;        // 16 waves (wave32)
// LDS strides (bf16 elements), padded, 16B-aligned rows
constexpr int XS  = Hc + 8;          // x tile     [MT][XS]
constexpr int WSB = KS + 8;          // w_in/w_v   [IC n][WSB k]   (transposed)
constexpr int HSB = IC + 8;          // h tile     [MT][HSB]
constexpr int WOS = 32 + 8;          // w_out      [Hc n][WOS k]   (transposed)

constexpr size_t SMEM_BYTES =
    (size_t)(MT * XS + 2 * IC * WSB + MT * HSB + Hc * WOS) * 2 /*bf16*/
    + MT * sizeof(int) + MT * sizeof(float);   // 301,568 B < 320 KB

struct U16x2 { uint4 a, b; };

// One-instruction bf16 pair pack: v_cvt_pk_bf16_f32 (src0 -> low, src1 -> high).
// Inline asm defeats the SLP vectorizer, which otherwise packs within each
// float4 and reassembles cross-register pairs with and/shl/or/mov_b16.
DI unsigned pack2(float lo, float hi) {
  unsigned r;
  asm("v_cvt_pk_bf16_f32 %0, %1, %2" : "=v"(r) : "v"(lo), "v"(hi));
  return r;
}

// A operand (16-bit, 16x32): lanes 0-15 row M=lane hold K {k0..+7, k0+16..+23},
// lanes 16-31 row M=lane-16 hold K {k0+8..+15, k0+24..+31}. Row-major LDS source.
DI v16bf lds_loadA(const bf16_t* base, int stride, int m0, int k0, int lane) {
  int row = m0 + (lane & 15);
  int kk  = k0 + ((lane & 16) >> 1);
  const bf16_t* p = base + row * stride + kk;
  U16x2 u;
  u.a = *(const uint4*)(p);
  u.b = *(const uint4*)(p + 16);
  return __builtin_bit_cast(v16bf, u);
}

// B operand (32x16): lanes 0-15 col N=lane hold K {k0..+7, k0+16..+23}, lanes 16-31
// col N=lane-16 hold K {k0+8..+15, k0+24..+31}. LDS stored transposed [n][k] (contiguous K).
DI v16bf lds_loadB(const bf16_t* base, int stride, int n0, int k0, int lane) {
  int col = n0 + (lane & 15);
  int kk  = k0 + ((lane & 16) >> 1);
  const bf16_t* p = base + col * stride + kk;
  U16x2 u; u.a = *(const uint4*)(p); u.b = *(const uint4*)(p + 16);
  return __builtin_bit_cast(v16bf, u);
}

DI v8f wmma_bf16(v16bf a, v16bf b, v8f c) {
  return __builtin_amdgcn_wmma_f32_16x16x32_bf16(false, a, false, b, (short)0, c,
                                                 false, false);
}

DI float gelu_tanh(float x) {
  float u = 0.7978845608028654f * (x + 0.044715f * x * x * x);
  return 0.5f * x * (1.0f + tanhf(u));
}

// ---------------------------------------------------------------------------
__global__ void zero_kernel(float4* __restrict__ out, int n4, int* __restrict__ counts) {
  int i = blockIdx.x * blockDim.x + threadIdx.x;
  if (i < n4) out[i] = make_float4(0.f, 0.f, 0.f, 0.f);
  if (blockIdx.x == 0 && threadIdx.x < Ec) counts[threadIdx.x] = 0;
}

// ---------------------------------------------------------------------------
// One wave per token: logits over E=8 experts, top-2 + softmax, bucket scatter.
__global__ __launch_bounds__(256)
void router_kernel(const float* __restrict__ hidden, const float* __restrict__ gate_w,
                   float* __restrict__ logits, int* __restrict__ counts,
                   int* __restrict__ btok, float* __restrict__ bw) {
  const int lane = threadIdx.x & 31;
  const int wave = threadIdx.x >> 5;
  const int t = blockIdx.x * 8 + wave;
  const float* xr = hidden + (size_t)t * Hc;

  float acc[Ec];
#pragma unroll
  for (int e = 0; e < Ec; ++e) acc[e] = 0.f;

  for (int h = lane; h < Hc; h += 32) {
    float xv = xr[h];
    const float4 g0 = *(const float4*)(gate_w + h * Ec);
    const float4 g1 = *(const float4*)(gate_w + h * Ec + 4);
    acc[0] += xv * g0.x; acc[1] += xv * g0.y; acc[2] += xv * g0.z; acc[3] += xv * g0.w;
    acc[4] += xv * g1.x; acc[5] += xv * g1.y; acc[6] += xv * g1.z; acc[7] += xv * g1.w;
  }
#pragma unroll
  for (int e = 0; e < Ec; ++e) {
#pragma unroll
    for (int off = 16; off > 0; off >>= 1) acc[e] += __shfl_xor(acc[e], off, 32);
  }
  if (lane == 0) {
    float* lo = logits + (size_t)t * Ec;
#pragma unroll
    for (int e = 0; e < Ec; ++e) lo[e] = acc[e];
    int e0 = 0; float m0 = acc[0];
#pragma unroll
    for (int e = 1; e < Ec; ++e) if (acc[e] > m0) { m0 = acc[e]; e0 = e; }
    int e1 = (e0 == 0) ? 1 : 0; float m1 = -3.4e38f;
#pragma unroll
    for (int e = 0; e < Ec; ++e) if (e != e0 && acc[e] > m1) { m1 = acc[e]; e1 = e; }
    float p1 = expf(m1 - m0);
    float inv = 1.f / (1.f + p1);
    float w0 = inv, w1 = p1 * inv;
    int i0 = atomicAdd(&counts[e0], 1);
    btok[e0 * Tc + i0] = t; bw[e0 * Tc + i0] = w0;
    int i1 = atomicAdd(&counts[e1], 1);
    btok[e1 * Tc + i1] = t; bw[e1 * Tc + i1] = w1;
  }
}

// ---------------------------------------------------------------------------
// Fused expert MLP over a gathered tile of 64 tokens for one expert.
__global__ __launch_bounds__(NTHREADS)
void moe_kernel(const float* __restrict__ hidden, const float* __restrict__ w_in,
                const float* __restrict__ w_v, const float* __restrict__ w_out,
                float* __restrict__ out_final, const int* __restrict__ counts,
                const int* __restrict__ btok, const float* __restrict__ bw) {
  const int tile = blockIdx.x;
  const int e = blockIdx.y;
  const int cnt = counts[e];
  if (tile * MT >= cnt) return;

  extern __shared__ char smem_raw[];
  bf16_t* xs    = (bf16_t*)smem_raw;           // [MT][XS]
  bf16_t* wsin  = xs + MT * XS;                // [IC][WSB]  ([n][k])
  bf16_t* wsv   = wsin + IC * WSB;             // [IC][WSB]
  bf16_t* hs    = wsv + IC * WSB;              // [MT][HSB]
  bf16_t* wo    = hs + MT * HSB;               // [Hc][WOS]  ([n][k])
  int*    tok_s = (int*)(wo + Hc * WOS);
  float*  wts_s = (float*)(tok_s + MT);
  unsigned* wsin32 = (unsigned*)wsin;          // packed k-pair view
  unsigned* wsv32  = (unsigned*)wsv;
  unsigned* wo32   = (unsigned*)wo;
  constexpr int WSU = WSB / 2;                 // 68 uints per n row
  constexpr int WOU = WOS / 2;                 // 20 uints per n row

  const int tid  = threadIdx.x;
  const int lane = tid & 31;
  const int wave = tid >> 5;                   // 16 waves

  if (tid < MT) {
    int idx = tile * MT + tid;
    int tk = 0; float w = 0.f;
    if (idx < cnt) { tk = btok[e * Tc + idx]; w = bw[e * Tc + idx]; }
    tok_s[tid] = tk; wts_s[tid] = w;
  }
  __syncthreads();

  // gather x tile -> LDS (fp32->bf16); batch 4 loads to keep them in flight
#pragma unroll
  for (int it = 0; it < 8; ++it) {
    float4 r[4];
    int rowv[4], c4v[4];
#pragma unroll
    for (int u = 0; u < 4; ++u) {
      int i = tid + (it * 4 + u) * NTHREADS;   // 32 float4/thread total
      rowv[u] = i >> 8; c4v[u] = i & 255;
      r[u] = *(const float4*)(hidden + (size_t)tok_s[rowv[u]] * Hc + c4v[u] * 4);
    }
#pragma unroll
    for (int u = 0; u < 4; ++u) {
      uint2 pk; pk.x = pack2(r[u].x, r[u].y); pk.y = pack2(r[u].z, r[u].w);
      *(uint2*)(xs + rowv[u] * XS + c4v[u] * 4) = pk;
    }
  }

  const float* w_in_e  = w_in  + (size_t)e * Hc * Ic;
  const float* w_v_e   = w_v   + (size_t)e * Hc * Ic;
  const float* w_out_e = w_out + (size_t)e * Ic * Hc;

  // phase-1 mapping: wave -> (mt, nt0/nt1) pair of (G,V) tiles; 32 pairs total
  const int mt1    = wave >> 2;                // 0..3
  const int nt0    = (wave & 3) * 2;           // 0,2,4,6
  const int nt1    = nt0 + 1;
  const int rowAdd = (lane & 16) >> 1;         // C-layout: +8 rows upper half
  const int colL   = lane & 15;

  v8f accY[4][4];
#pragma unroll
  for (int a = 0; a < 4; ++a)
#pragma unroll
    for (int b = 0; b < 4; ++b) accY[a][b] = v8f{};

  for (int c = 0; c < Ic / IC; ++c) {          // 32 chunks over I
    const int n0 = c * IC;
    v8f accG[2] = {v8f{}, v8f{}}, accV[2] = {v8f{}, v8f{}};

    // ---- phase 1: G = x @ w_in[:,chunk], V = x @ w_v[:,chunk] ----
    for (int r = 0; r < Hc / KS; ++r) {        // 8 staging rounds of K=128
      const int k0g = r * KS;
      __syncthreads();
      // stage [128k x 128n] of w_in & w_v, fp32->bf16 transposed, packed k-pairs
#pragma unroll
      for (int it = 0; it < 4; ++it) {
        int j  = tid + it * NTHREADS;          // 2048 micro-tiles (2k x 4n)
        int kp = j >> 5;                       // 0..63
        int n4 = j & 31;                       // 0..31
        const float* p0 = w_in_e + (size_t)(k0g + kp * 2) * Ic + n0 + n4 * 4;
        const float* p1 = w_v_e  + (size_t)(k0g + kp * 2) * Ic + n0 + n4 * 4;
        if (it == 0 && r + 1 < Hc / KS) {
          __builtin_prefetch(p0 + (size_t)KS * Ic, 0, 0);
          __builtin_prefetch(p1 + (size_t)KS * Ic, 0, 0);
        }
        float4 a0 = *(const float4*)(p0);
        float4 a1 = *(const float4*)(p0 + Ic);
        float4 b0 = *(const float4*)(p1);
        float4 b1 = *(const float4*)(p1 + Ic);
        int nb = n4 * 4;
        wsin32[(nb + 0) * WSU + kp] = pack2(a0.x, a1.x);
        wsin32[(nb + 1) * WSU + kp] = pack2(a0.y, a1.y);
        wsin32[(nb + 2) * WSU + kp] = pack2(a0.z, a1.z);
        wsin32[(nb + 3) * WSU + kp] = pack2(a0.w, a1.w);
        wsv32 [(nb + 0) * WSU + kp] = pack2(b0.x, b1.x);
        wsv32 [(nb + 1) * WSU + kp] = pack2(b0.y, b1.y);
        wsv32 [(nb + 2) * WSU + kp] = pack2(b0.z, b1.z);
        wsv32 [(nb + 3) * WSU + kp] = pack2(b0.w, b1.w);
      }
      __syncthreads();
#pragma unroll
      for (int kk = 0; kk < KS / 32; ++kk) {   // 4 K-steps, no barriers
        int kl = kk * 32;
        v16bf A   = lds_loadA(xs, XS, mt1 * 16, k0g + kl, lane);
        v16bf Bg0 = lds_loadB(wsin, WSB, nt0 * 16, kl, lane);
        v16bf Bg1 = lds_loadB(wsin, WSB, nt1 * 16, kl, lane);
        v16bf Bv0 = lds_loadB(wsv,  WSB, nt0 * 16, kl, lane);
        v16bf Bv1 = lds_loadB(wsv,  WSB, nt1 * 16, kl, lane);
        accG[0] = wmma_bf16(A, Bg0, accG[0]);
        accG[1] = wmma_bf16(A, Bg1, accG[1]);
        accV[0] = wmma_bf16(A, Bv0, accV[0]);
        accV[1] = wmma_bf16(A, Bv1, accV[1]);
      }
    }

    // ---- activation: h = gelu_tanh(G) * V -> LDS bf16 ----
#pragma unroll
    for (int j = 0; j < 2; ++j) {
      int nt = nt0 + j;
#pragma unroll
      for (int rr = 0; rr < 8; ++rr) {
        float g = gelu_tanh(accG[j][rr]);
        float h = g * accV[j][rr];
        hs[(mt1 * 16 + rr + rowAdd) * HSB + nt * 16 + colL] = (bf16_t)h;
      }
    }

    // ---- phase 2: y += h @ w_out[chunk,:] ----
#pragma unroll
    for (int s = 0; s < IC / 32; ++s) {        // 4 K-steps of 32 over IC
      __syncthreads();                         // prior wo reads + hs writes done
      // stage [32k x 1024n] of w_out, micro-tiles 2k x 4n, 2 tiles per iter
#pragma unroll
      for (int it = 0; it < 4; ++it) {
        int j0 = tid + (it * 2 + 0) * NTHREADS;
        int j1 = tid + (it * 2 + 1) * NTHREADS;
        int kp0 = j0 >> 8, n40 = j0 & 255;
        int kp1 = j1 >> 8, n41 = j1 & 255;
        const float* p0 = w_out_e + (size_t)(n0 + s * 32 + kp0 * 2) * Hc + n40 * 4;
        const float* p1 = w_out_e + (size_t)(n0 + s * 32 + kp1 * 2) * Hc + n41 * 4;
        float4 a0 = *(const float4*)(p0);
        float4 a1 = *(const float4*)(p0 + Hc);
        float4 b0 = *(const float4*)(p1);
        float4 b1 = *(const float4*)(p1 + Hc);
        int nb0 = n40 * 4, nb1 = n41 * 4;
        wo32[(nb0 + 0) * WOU + kp0] = pack2(a0.x, a1.x);
        wo32[(nb0 + 1) * WOU + kp0] = pack2(a0.y, a1.y);
        wo32[(nb0 + 2) * WOU + kp0] = pack2(a0.z, a1.z);
        wo32[(nb0 + 3) * WOU + kp0] = pack2(a0.w, a1.w);
        wo32[(nb1 + 0) * WOU + kp1] = pack2(b0.x, b1.x);
        wo32[(nb1 + 1) * WOU + kp1] = pack2(b0.y, b1.y);
        wo32[(nb1 + 2) * WOU + kp1] = pack2(b0.z, b1.z);
        wo32[(nb1 + 3) * WOU + kp1] = pack2(b0.w, b1.w);
      }
      __syncthreads();
      v16bf a0 = lds_loadA(hs, HSB, 0,  s * 32, lane);
      v16bf a1 = lds_loadA(hs, HSB, 16, s * 32, lane);
      v16bf a2 = lds_loadA(hs, HSB, 32, s * 32, lane);
      v16bf a3 = lds_loadA(hs, HSB, 48, s * 32, lane);
      v16bf bcur = lds_loadB(wo, WOS, wave * 64, 0, lane);
#pragma unroll
      for (int nt = 0; nt < 4; ++nt) {
        v16bf bnext = bcur;
        if (nt < 3) bnext = lds_loadB(wo, WOS, wave * 64 + (nt + 1) * 16, 0, lane);
        accY[0][nt] = wmma_bf16(a0, bcur, accY[0][nt]);
        accY[1][nt] = wmma_bf16(a1, bcur, accY[1][nt]);
        accY[2][nt] = wmma_bf16(a2, bcur, accY[2][nt]);
        accY[3][nt] = wmma_bf16(a3, bcur, accY[3][nt]);
        bcur = bnext;
      }
    }
  }

  // ---- scaled accumulate into shared output (token may be in 2 buckets) ----
#pragma unroll
  for (int mt = 0; mt < 4; ++mt) {
#pragma unroll
    for (int rr = 0; rr < 8; ++rr) {
      int row = mt * 16 + rr + rowAdd;
      int tk = tok_s[row];
      float w = wts_s[row];
      float* dst = out_final + (size_t)tk * Hc + wave * 64 + colL;
#pragma unroll
      for (int nt = 0; nt < 4; ++nt) {
        unsafeAtomicAdd(dst + nt * 16, w * accY[mt][nt][rr]);
      }
    }
  }
}

// ---------------------------------------------------------------------------
extern "C" void kernel_launch(void* const* d_in, const int* in_sizes, int n_in,
                              void* d_out, int out_size, void* d_ws, size_t ws_size,
                              hipStream_t stream) {
  (void)in_sizes; (void)n_in; (void)out_size; (void)ws_size;
  const float* hidden = (const float*)d_in[0];
  const float* gate_w = (const float*)d_in[1];
  const float* w_in   = (const float*)d_in[2];
  const float* w_v    = (const float*)d_in[3];
  const float* w_out  = (const float*)d_in[4];

  float* out    = (float*)d_out;              // [T, H] final
  float* logits = out + (size_t)Tc * Hc;      // [T, E] router logits

  int*   counts = (int*)d_ws;                 // 16 ints (8 used)
  int*   btok   = counts + 16;                // [E][T]
  float* bw     = (float*)(btok + Ec * Tc);   // [E][T]

  int n4 = (Tc * Hc) / 4;
  zero_kernel<<<(n4 + 255) / 256, 256, 0, stream>>>((float4*)out, n4, counts);
  router_kernel<<<Tc / 8, 256, 0, stream>>>(hidden, gate_w, logits, counts, btok, bw);
  dim3 grid(Tc / MT, Ec);
  moe_kernel<<<grid, NTHREADS, SMEM_BYTES, stream>>>(hidden, w_in, w_v, w_out, out,
                                                     counts, btok, bw);
}
